// Decoder_12266426597950
// MI455X (gfx1250) — matrix operand order repaired and checked by reading, not compiled, not run on hardware
//
#include <hip/hip_runtime.h>
#include <stdint.h>

// ---- problem dims (match reference) ----
#define Bx 256
#define Tt 512
#define Ff 128
#define Uu 256
#define Gg 1024   // 4*U
#define Hh 64
#define NTHREADS 512   // 16 waves

typedef __bf16 v16bf __attribute__((ext_vector_type(16)));
typedef __bf16 v8bf  __attribute__((ext_vector_type(8)));
typedef float  v8f   __attribute__((ext_vector_type(8)));
typedef float  v4f   __attribute__((ext_vector_type(4)));
typedef unsigned int v4u __attribute__((ext_vector_type(4)));

// float -> bf16 bits, round-to-nearest-even
__device__ __forceinline__ unsigned int f2bf(float f) {
  unsigned int u = __float_as_uint(f);
  u += 0x7FFFu + ((u >> 16) & 1u);
  return u >> 16;
}

// ---------------- prep kernels ----------------
__global__ void cvt_kernel(const float* __restrict__ src,
                           unsigned short* __restrict__ dst, long n) {
  long i = (long)blockIdx.x * blockDim.x + threadIdx.x;
  if (i < n) dst[i] = (unsigned short)f2bf(src[i]);
}

// src[rows][cols] fp32 -> dst[cols][rows] bf16 (N-major weights for B fragments)
__global__ void cvtT_kernel(const float* __restrict__ src,
                            unsigned short* __restrict__ dst, int rows, int cols) {
  long i = (long)blockIdx.x * blockDim.x + threadIdx.x;
  long n = (long)rows * cols;
  if (i < n) {
    int r = (int)(i / cols);
    int c = (int)(i % cols);
    dst[(long)c * rows + r] = (unsigned short)f2bf(src[i]);
  }
}

// One 16x16x32 bf16 WMMA step.
// A layout (16-bit, 16x32): lane -> row = lane&15; half = lane>>4;
//   lo 8 elems at K = base+half*16 .. +7 ; hi 8 at +8 .. +15
// B layout (32x16): lane -> col = lane&15; lanes 0-15 K=0..15, lanes 16-31 K=16..31
__device__ __forceinline__ v8f wmma_step(const unsigned short* __restrict__ aptr,
                                         int a_lo, int a_hi,
                                         const unsigned short* __restrict__ bptr,
                                         v8f acc) {
  v16bf a, b;
  *((v8bf*)&a)       = *(const v8bf*)(aptr + a_lo);
  *(((v8bf*)&a) + 1) = *(const v8bf*)(aptr + a_hi);
  *((v8bf*)&b)       = *(const v8bf*)(bptr);
  *(((v8bf*)&b) + 1) = *(const v8bf*)(bptr + 8);
  return __builtin_amdgcn_wmma_f32_16x16x32_bf16(false, a, false, b,
                                                 (short)0, acc, false, false);
}

// ---------------- input-projection GEMM (fully parallel, whole chip) ----------
// zx[131072,1024] = xb[131072,128] @ kernel[128,1024] + bias   (bf16 out)
// Grid: 8192 WGs x 512 threads; each WG one 16-row stripe, wave -> 4 N-tiles.
__launch_bounds__(NTHREADS, 1)
__global__ void xproj_kernel(const unsigned short* __restrict__ xb,  // [131072][128]
                             const unsigned short* __restrict__ kT,  // [1024][128]
                             const float* __restrict__ bias,         // [1024]
                             unsigned short* __restrict__ zx)        // [131072][1024]
{
  const int tid  = threadIdx.x;
  const int lane = tid & 31;
  const int wave = tid >> 5;
  const int l15  = lane & 15;
  const int half = lane >> 4;
  const size_t row0 = (size_t)blockIdx.x * 16;

  const unsigned short* arow = xb + (row0 + l15) * Ff;

#pragma unroll 1
  for (int nt4 = 0; nt4 < 4; ++nt4) {
    const int nt  = wave * 4 + nt4;
    const int col = nt * 16 + l15;
    const float bv = bias[col];
    v8f acc = {bv, bv, bv, bv, bv, bv, bv, bv};
    const unsigned short* bW = kT + (size_t)col * Ff;
#pragma unroll
    for (int kc = 0; kc < 4; ++kc) {
      const int alo = kc * 32 + half * 16;
      acc = wmma_step(arow, alo, alo + 8, bW + kc * 32 + half * 16, acc);
    }
#pragma unroll
    for (int j = 0; j < 8; ++j)
      zx[(row0 + half * 8 + j) * Gg + col] = (unsigned short)f2bf(acc[j]);
  }
}

// ---------------- fused recurrent LSTM + head kernel ----------------
// Grid: 16 workgroups, each owns 16 batch rows, 512 threads = 16 waves.
__launch_bounds__(NTHREADS, 1)
__global__ void lstm_head_fused(
    const float* __restrict__ b_out,        // [64]
    const unsigned short* __restrict__ zx,  // [B*T][1024] bf16: x@W + bias
    const unsigned short* __restrict__ rT,  // [1024][256] bf16 (rec_kernel^T)
    const unsigned short* __restrict__ wT,  // [64][T*U]   bf16 (w_out^T)
    float* __restrict__ out)                // [B][64]
{
  __shared__ float zbuf[16][Gg];                        // 64 KB recurrent pre-acts
  __shared__ float cbuf[16][Uu];                        // 16 KB cell state
  __shared__ __align__(16) unsigned short hbuf[16][Uu]; // 8 KB hidden state (bf16)

  const int tid  = threadIdx.x;
  const int lane = tid & 31;
  const int wave = tid >> 5;        // 0..15
  const int l15  = lane & 15;
  const int half = lane >> 4;       // 0|1
  const int b0   = blockIdx.x * 16; // batch row base

  for (int i = tid; i < 16 * Uu; i += NTHREADS) {
    (&cbuf[0][0])[i] = 0.0f;
    (&hbuf[0][0])[i] = 0;
  }

  // persistent head accumulator: wave -> (N tile = wave&3, K quarter = wave>>2)
  v8f hacc = {0.f,0.f,0.f,0.f,0.f,0.f,0.f,0.f};
  const int ntile_h = wave & 3;
  const int kq      = wave >> 2;    // 0..3, 64 K-elements each

  // gate-phase assignment: row gr, units gu0..gu0+7
  const int gr  = tid >> 5;
  const int gu0 = (tid & 31) * 8;
  const unsigned short* zxbase = zx + (size_t)(b0 + gr) * Tt * Gg + gu0;

  __syncthreads();

  // preload zx for t=0 (i,f,g,o blocks: 8 bf16 each)
  v4u pz[4];
#pragma unroll
  for (int g = 0; g < 4; ++g) pz[g] = *(const v4u*)(zxbase + g * Uu);

#pragma unroll 1
  for (int t = 0; t < Tt; ++t) {
    // ---- zbuf[16,1024] = h @ R : 4 col tiles per wave, K=256 ----
#pragma unroll 1
    for (int nt4 = 0; nt4 < 4; ++nt4) {
      const int nt  = wave * 4 + nt4;   // column tile 0..63
      const int col = nt * 16 + l15;
      v8f acc = {0.f,0.f,0.f,0.f,0.f,0.f,0.f,0.f};
      const unsigned short* bR   = rT + (size_t)col * Uu;
      const unsigned short* hrow = &hbuf[l15][0];
#pragma unroll
      for (int kc = 0; kc < 8; ++kc) {
        const int alo = kc * 32 + half * 16;
        acc = wmma_step(hrow, alo, alo + 8, bR + kc * 32 + half * 16, acc);
      }
#pragma unroll
      for (int j = 0; j < 8; ++j)
        zbuf[half * 8 + j][col] = acc[j];
    }

    // issue next step's zx loads; consumed after next barrier pair -> a full
    // step of WMMA work hides their HBM/L2 latency.
    v4u pznext[4] = {pz[0], pz[1], pz[2], pz[3]};
    if (t + 1 < Tt) {
      const unsigned short* zxt = zxbase + (size_t)(t + 1) * Gg;
#pragma unroll
      for (int g = 0; g < 4; ++g) pznext[g] = *(const v4u*)(zxt + g * Uu);
    }
    __syncthreads();

    // ---- gates: z = (h@R) + zx ; 8 units per thread, vectorized LDS ----
    {
      float zi[8], zf[8], zg[8], zo[8], cc[8], hh[8];
      *(v4f*)&zi[0] = *(const v4f*)&zbuf[gr][gu0];
      *(v4f*)&zi[4] = *(const v4f*)&zbuf[gr][gu0 + 4];
      *(v4f*)&zf[0] = *(const v4f*)&zbuf[gr][Uu + gu0];
      *(v4f*)&zf[4] = *(const v4f*)&zbuf[gr][Uu + gu0 + 4];
      *(v4f*)&zg[0] = *(const v4f*)&zbuf[gr][2 * Uu + gu0];
      *(v4f*)&zg[4] = *(const v4f*)&zbuf[gr][2 * Uu + gu0 + 4];
      *(v4f*)&zo[0] = *(const v4f*)&zbuf[gr][3 * Uu + gu0];
      *(v4f*)&zo[4] = *(const v4f*)&zbuf[gr][3 * Uu + gu0 + 4];
      *(v4f*)&cc[0] = *(const v4f*)&cbuf[gr][gu0];
      *(v4f*)&cc[4] = *(const v4f*)&cbuf[gr][gu0 + 4];
      // unpack zx (bf16 pairs) and add
#pragma unroll
      for (int p = 0; p < 4; ++p) {
        zi[2*p]   += __uint_as_float(pz[0][p] << 16);
        zi[2*p+1] += __uint_as_float(pz[0][p] & 0xFFFF0000u);
        zf[2*p]   += __uint_as_float(pz[1][p] << 16);
        zf[2*p+1] += __uint_as_float(pz[1][p] & 0xFFFF0000u);
        zg[2*p]   += __uint_as_float(pz[2][p] << 16);
        zg[2*p+1] += __uint_as_float(pz[2][p] & 0xFFFF0000u);
        zo[2*p]   += __uint_as_float(pz[3][p] << 16);
        zo[2*p+1] += __uint_as_float(pz[3][p] & 0xFFFF0000u);
      }
#pragma unroll
      for (int e = 0; e < 8; ++e) {
        const float ig = __builtin_amdgcn_rcpf(1.0f + __expf(-zi[e]));
        const float fg = __builtin_amdgcn_rcpf(1.0f + __expf(-zf[e]));
        const float gg = zg[e] * __builtin_amdgcn_rcpf(1.0f + __expf(-zg[e]));
        const float og = __builtin_amdgcn_rcpf(1.0f + __expf(-zo[e]));
        const float c  = fg * cc[e] + ig * gg;
        cc[e] = c;
        hh[e] = og * (c * __builtin_amdgcn_rcpf(1.0f + __expf(-c)));
      }
      *(v4f*)&cbuf[gr][gu0]     = *(const v4f*)&cc[0];
      *(v4f*)&cbuf[gr][gu0 + 4] = *(const v4f*)&cc[4];
      v4u hp;
#pragma unroll
      for (int p = 0; p < 4; ++p)
        hp[p] = f2bf(hh[2 * p]) | (f2bf(hh[2 * p + 1]) << 16);
      *(v4u*)&hbuf[gr][gu0] = hp;   // single ds_store_b128
    }
    __syncthreads();

    // ---- fused head: hacc += h_t @ w_out_t (4 N-tiles x 4 K-quarters) ----
    {
      const unsigned short* hrow = &hbuf[l15][kq * 64];
      const unsigned short* bptr = wT + (size_t)(ntile_h * 16 + l15) * (size_t)(Tt * Uu)
                                      + (size_t)t * Uu + kq * 64;
#pragma unroll
      for (int kc = 0; kc < 2; ++kc) {
        const int alo = kc * 32 + half * 16;
        hacc = wmma_step(hrow, alo, alo + 8, bptr + kc * 32 + half * 16, hacc);
      }
    }
    pz[0] = pznext[0]; pz[1] = pznext[1]; pz[2] = pznext[2]; pz[3] = pznext[3];
    // next iteration's z-GEMM reads the same hbuf; hbuf is only rewritten
    // after the post-z __syncthreads, so no extra barrier needed here.
  }

  // ---- reduce K-quarter head partials into kq==0 waves, add b_out ----
  float* part = &zbuf[0][0];  // reuse LDS: [wave][16*16]
  __syncthreads();
  if (kq != 0) {
#pragma unroll
    for (int j = 0; j < 8; ++j)
      part[wave * 256 + (half * 8 + j) * 16 + l15] = hacc[j];
  }
  __syncthreads();
  if (kq == 0) {
    const int col = ntile_h * 16 + l15;
    const float bo = b_out[col];
#pragma unroll
    for (int j = 0; j < 8; ++j) {
      float v = hacc[j];
#pragma unroll
      for (int q = 1; q < 4; ++q)
        v += part[(q * 4 + ntile_h) * 256 + (half * 8 + j) * 16 + l15];
      out[(size_t)(b0 + half * 8 + j) * Hh + col] = v + bo;
    }
  }
}

// ---------------- launch ----------------
extern "C" void kernel_launch(void* const* d_in, const int* in_sizes, int n_in,
                              void* d_out, int out_size, void* d_ws, size_t ws_size,
                              hipStream_t stream) {
  const float* x     = (const float*)d_in[0];  // [256,512,128]
  const float* kern  = (const float*)d_in[1];  // [128,1024]
  const float* rec   = (const float*)d_in[2];  // [256,1024]
  const float* bias  = (const float*)d_in[3];  // [1024]
  const float* w_out = (const float*)d_in[4];  // [131072,64]
  const float* b_out = (const float*)d_in[5];  // [64]
  float* out = (float*)d_out;                  // [256,64]

  char* ws = (char*)d_ws;
  size_t off = 0;
  auto wsalloc = [&](size_t bytes) -> void* {
    void* p = ws + off;
    off = (off + bytes + 255) & ~(size_t)255;
    return p;
  };
  unsigned short* xb = (unsigned short*)wsalloc((size_t)Bx * Tt * Ff * 2); // 32 MB
  unsigned short* kT = (unsigned short*)wsalloc((size_t)Gg * Ff * 2);      // 256 KB
  unsigned short* rT = (unsigned short*)wsalloc((size_t)Gg * Uu * 2);      // 512 KB
  unsigned short* wT = (unsigned short*)wsalloc((size_t)Hh * Tt * Uu * 2); // 16 MB
  unsigned short* zx = (unsigned short*)wsalloc((size_t)Bx * Tt * Gg * 2); // 256 MB

  const long nx = (long)Bx * Tt * Ff;
  cvt_kernel<<<(unsigned)((nx + 255) / 256), 256, 0, stream>>>(x, xb, nx);
  const long nk = (long)Ff * Gg;
  cvtT_kernel<<<(unsigned)((nk + 255) / 256), 256, 0, stream>>>(kern, kT, Ff, Gg);
  const long nr = (long)Uu * Gg;
  cvtT_kernel<<<(unsigned)((nr + 255) / 256), 256, 0, stream>>>(rec, rT, Uu, Gg);
  const long nw = (long)Tt * Uu * Hh;
  cvtT_kernel<<<(unsigned)((nw + 255) / 256), 256, 0, stream>>>(w_out, wT, Tt * Uu, Hh);

  // input projection over all (b,t) rows — fully parallel WMMA GEMM
  xproj_kernel<<<(Bx * Tt) / 16, NTHREADS, 0, stream>>>(xb, kT, bias, zx);

  // serial recurrence + fused dense head
  lstm_head_fused<<<Bx / 16, NTHREADS, 0, stream>>>(b_out, zx, rT, wT, out);
}
